// MixtralBLockSparseTop2MLP_HQQ_18279380812578
// MI455X (gfx1250) — matrix operand order repaired and checked by reading, not compiled, not run on hardware
//
#include <hip/hip_runtime.h>
#include <hip/hip_bf16.h>

// ---------------------------------------------------------------------------
// Mixtral expert MLP, HQQ 4-bit (int32-stored) weights.
//   gate = x @ W1^T ; up = x @ W3^T ; inter = silu(gate)*up ; out = inter @ W2^T
// H=4096, F=14336, T=512, GS=64.
//
// Memory-bound: weights are 3*H*F*4B = 704MB -> ~30us at 23.3 TB/s. Every
// weight element is read from HBM and dequantized EXACTLY ONCE. Weight bytes
// are streamed global->LDS with GLOBAL_LOAD_ASYNC_TO_LDS_B128 (ASYNCcnt,
// double-buffered) so the HBM stream overlaps the WMMA phase; A-fragment
// LOADcnt waits never drain the weight prefetch. x / inter are f16 and
// L2-resident (192MB L2), so their per-block re-reads are cheap.
// ---------------------------------------------------------------------------

#define H_DIM 4096
#define F_DIM 14336
#define T_DIM 512
#define GS    64

#define FT    32            // output features per block
#define KC    64            // K chunk per pipeline stage
#define LBST  (KC + 8)      // f16 panel row stride in halves (16B-aligned rows)
#define NTT   8             // 16-token tiles per wave (8*16 = 128 tokens/wave)

typedef __attribute__((ext_vector_type(16))) _Float16 v16h;
typedef __attribute__((ext_vector_type(8)))  _Float16 v8h;
typedef __attribute__((ext_vector_type(8)))  float    v8f;
typedef __attribute__((ext_vector_type(4)))  int      v4i;
typedef __attribute__((ext_vector_type(4)))  float    v4f;

__device__ __forceinline__ unsigned int pack2h(float a, float b) {
    union { _Float16 h[2]; unsigned int u; } u_;
    u_.h[0] = (_Float16)a; u_.h[1] = (_Float16)b;
    return u_.u;
}

__device__ __forceinline__ unsigned lds_off(const void* p) {
    return (unsigned)(size_t)p;     // low 32 bits of generic ptr = LDS byte offset
}

// Async DMA: 16B global -> LDS, tracked by ASYNCcnt (not LOADcnt).
__device__ __forceinline__ void async_copy_b128(unsigned dst_lds, const void* src) {
    asm volatile("global_load_async_to_lds_b128 %0, %1, off"
                 :: "v"(dst_lds), "v"(src) : "memory");
}

__device__ __forceinline__ void wait_async_le4() {
    asm volatile("s_wait_asynccnt 4" ::: "memory");
}
__device__ __forceinline__ void wait_async_le2() {
    asm volatile("s_wait_asynccnt 2" ::: "memory");
}
__device__ __forceinline__ void wait_async_le0() {
    asm volatile("s_wait_asynccnt 0" ::: "memory");
}

// A fragment (16x32 f16) from global f16 matrix [rows x ld].
// ISA 7.12.2: lane<16 -> row=lane,   K = kk+[0..7] and kk+[16..23]
//             lane>=16 -> row=lane-16, K = kk+[8..15] and kk+[24..31]
__device__ __forceinline__ v16h load_a_global(const _Float16* __restrict__ base,
                                              int ld, int row0, int kk, int lane) {
    const int m  = lane & 15;
    const int kb = (lane < 16) ? 0 : 8;
    const _Float16* p = base + (size_t)(row0 + m) * ld + kk + kb;
    v8h lo = *(const v8h*)(p);
    v8h hi = *(const v8h*)(p + 16);
    v16h a;
#pragma unroll
    for (int e = 0; e < 8; ++e) { a[e] = lo[e]; a[8 + e] = hi[e]; }
    return a;
}

// B fragment (32x16 f16) from LDS dequantized panel [32][LBST].
// lane<16: column=lane, K = kk+[0..7] and kk+[16..23]; lane>=16: column=lane-16, K = kk+[8..15] and kk+[24..31]
__device__ __forceinline__ v16h load_b_lds(const _Float16* __restrict__ panel,
                                           int fcol0, int kk, int lane) {
    const int col  = fcol0 + (lane & 15);
    const int koff = kk + ((lane < 16) ? 0 : 8);
    const _Float16* p = panel + col * LBST + koff;
    v8h lo = *(const v8h*)(p);
    v8h hi = *(const v8h*)(p + 16);
    v16h b;
#pragma unroll
    for (int e = 0; e < 8; ++e) { b[e] = lo[e]; b[8 + e] = hi[e]; }
    return b;
}

__device__ __forceinline__ v8f wmma_f16(v16h a, v16h b, v8f c) {
    return __builtin_amdgcn_wmma_f32_16x16x32_f16(
        false, a, false, b, (short)0, c, false, false);
}

// Dequant one 4-int segment (already in LDS) into the f16 panel.
__device__ __forceinline__ void dequant4(_Float16* __restrict__ dst,
                                         const int* __restrict__ q_lds,
                                         const float* __restrict__ s4,
                                         const float* __restrict__ z4) {
    v4i q  = *(const v4i*)q_lds;    // ds_load_b128
    v4f sc = *(const v4f*)s4;
    v4f zr = *(const v4f*)z4;
    float w0 = ((float)q[0] - zr[0]) * sc[0];
    float w1 = ((float)q[1] - zr[1]) * sc[1];
    float w2 = ((float)q[2] - zr[2]) * sc[2];
    float w3 = ((float)q[3] - zr[3]) * sc[3];
    *(unsigned int*)(dst)     = pack2h(w0, w1);
    *(unsigned int*)(dst + 2) = pack2h(w2, w3);
}

// ===========================================================================
// Kernel 0: x f32 -> f16
// ===========================================================================
__global__ __launch_bounds__(256)
void cvt_x_kernel(const float* __restrict__ x, _Float16* __restrict__ xh) {
    size_t i = ((size_t)blockIdx.x * 256 + threadIdx.x) * 4;
    float4 v = *(const float4*)(x + i);
    *(unsigned int*)(xh + i)     = pack2h(v.x, v.y);
    *(unsigned int*)(xh + i + 2) = pack2h(v.z, v.w);
}

// ===========================================================================
// Kernel 1: fused gate+up. Block = 512 tokens x 32 f. Grid = F/32 = 448.
// 8 waves: wr = wv>>1 -> 128-token quarter; wc = wv&1 -> 16-f half.
// Pipeline: async raw-q copy (chunk c+1) || dequant+WMMA (chunk c).
// ===========================================================================
__global__ __launch_bounds__(256)
void hqq_gateup_kernel(const _Float16* __restrict__ xh,
                       const int*   __restrict__ w1q,
                       const float* __restrict__ w1s,
                       const float* __restrict__ w1z,
                       const int*   __restrict__ w3q,
                       const float* __restrict__ w3s,
                       const float* __restrict__ w3z,
                       _Float16*    __restrict__ inter) {
    __shared__ __align__(16) int      rawQ[2][2][FT * KC];      // [buf][mat] 32KB
    __shared__ __align__(16) _Float16 ldsB[2][FT * LBST];       // [mat] 9.2KB

    const int tid  = threadIdx.x;
    const int lane = tid & 31;
    const int wv   = tid >> 5;
    const int wr   = wv >> 1;
    const int wc   = wv & 1;

    const int f0 = blockIdx.x * FT;
    const int g  = blockIdx.x >> 1;          // f0 / GS
    const int tw = wr * 128;

    v8f acc_g[NTT];
    v8f acc_u[NTT];
#pragma unroll
    for (int t = 0; t < NTT; ++t) { acc_g[t] = (v8f){}; acc_u[t] = (v8f){}; }

    // per-thread segment map: idx in [0, FT*KC/4); fl = idx/16, kloc = (idx%16)*4
    const int idx0 = tid;           // segment 0
    const int idx1 = tid + 256;     // segment 1

    // ---- prologue: async-stage chunk 0 -------------------------------------
#pragma unroll
    for (int r = 0; r < 2; ++r) {
        const int idx  = (r == 0) ? idx0 : idx1;
        const int fl   = idx >> 4;
        const int kloc = (idx & 15) << 2;
        const size_t goff = (size_t)(f0 + fl) * H_DIM + kloc;
        async_copy_b128(lds_off(&rawQ[0][0][idx * 4]), w1q + goff);
        async_copy_b128(lds_off(&rawQ[0][1][idx * 4]), w3q + goff);
    }

    const int nch = H_DIM / KC;              // 64 chunks
    for (int c = 0; c < nch; ++c) {
        const int cur = c & 1;
        const int k0  = c * KC;

        // ---- issue async copies for chunk c+1, then wait for chunk c -------
        if (c + 1 < nch) {
            const int nk0 = k0 + KC;
#pragma unroll
            for (int r = 0; r < 2; ++r) {
                const int idx  = (r == 0) ? idx0 : idx1;
                const int fl   = idx >> 4;
                const int kloc = (idx & 15) << 2;
                const size_t goff = (size_t)(f0 + fl) * H_DIM + nk0 + kloc;
                async_copy_b128(lds_off(&rawQ[cur ^ 1][0][idx * 4]), w1q + goff);
                async_copy_b128(lds_off(&rawQ[cur ^ 1][1][idx * 4]), w3q + goff);
            }
            wait_async_le4();                // in-order: chunk c's 4 copies done
        } else {
            wait_async_le0();
        }
        __syncthreads();                     // raw[cur] visible; f16 panel free

        // ---- dequant raw[cur] (LDS) -> f16 panels (LDS) --------------------
#pragma unroll
        for (int r = 0; r < 2; ++r) {
            const int idx  = (r == 0) ? idx0 : idx1;
            const int fl   = idx >> 4;
            const int kloc = (idx & 15) << 2;
            const size_t soff = (size_t)g * H_DIM + k0 + kloc;
            dequant4(&ldsB[0][fl * LBST + kloc], &rawQ[cur][0][idx * 4],
                     w1s + soff, w1z + soff);
            dequant4(&ldsB[1][fl * LBST + kloc], &rawQ[cur][1][idx * 4],
                     w3s + soff, w3z + soff);
        }
        __syncthreads();

        // ---- WMMA: each B fragment reused by 8 token tiles -----------------
#pragma unroll
        for (int kk = 0; kk < KC; kk += 32) {
            v16h b1 = load_b_lds(ldsB[0], wc * 16, kk, lane);
            v16h b3 = load_b_lds(ldsB[1], wc * 16, kk, lane);
#pragma unroll
            for (int t = 0; t < NTT; ++t) {
                v16h a = load_a_global(xh, H_DIM, tw + t * 16, k0 + kk, lane);
                acc_g[t] = wmma_f16(a, b1, acc_g[t]);
                acc_u[t] = wmma_f16(a, b3, acc_u[t]);
            }
        }
    }

    // ---- epilogue: silu(gate)*up -> f16 inter [T, F] -----------------------
    const int n     = lane & 15;
    const int mbase = (lane < 16) ? 0 : 8;
#pragma unroll
    for (int t = 0; t < NTT; ++t) {
#pragma unroll
        for (int i = 0; i < 8; ++i) {
            float gg = acc_g[t][i];
            float uu = acc_u[t][i];
            float sv = gg / (1.0f + __expf(-gg));
            int trow = tw + t * 16 + mbase + i;
            inter[(size_t)trow * F_DIM + f0 + wc * 16 + n] = (_Float16)(sv * uu);
        }
    }
}

// ===========================================================================
// Kernel 2: down projection. Block = 512 tokens x 32 h. Grid = H/32 = 128.
// ===========================================================================
__global__ __launch_bounds__(256)
void hqq_down_kernel(const _Float16* __restrict__ inter,
                     const int*      __restrict__ w2q,
                     const float*    __restrict__ w2s,
                     const float*    __restrict__ w2z,
                     float*          __restrict__ out) {
    __shared__ __align__(16) int      rawQ[2][FT * KC];         // 16KB
    __shared__ __align__(16) _Float16 ldsB[FT * LBST];          // 4.6KB

    const int tid  = threadIdx.x;
    const int lane = tid & 31;
    const int wv   = tid >> 5;
    const int wr   = wv >> 1;
    const int wc   = wv & 1;

    const int h0 = blockIdx.x * FT;
    const int g  = blockIdx.x >> 1;
    const int tw = wr * 128;

    v8f acc[NTT];
#pragma unroll
    for (int t = 0; t < NTT; ++t) acc[t] = (v8f){};

    const int idx0 = tid;
    const int idx1 = tid + 256;

#pragma unroll
    for (int r = 0; r < 2; ++r) {
        const int idx  = (r == 0) ? idx0 : idx1;
        const int fl   = idx >> 4;
        const int kloc = (idx & 15) << 2;
        async_copy_b128(lds_off(&rawQ[0][idx * 4]),
                        w2q + (size_t)(h0 + fl) * F_DIM + kloc);
    }

    const int nch = F_DIM / KC;              // 224 chunks
    for (int c = 0; c < nch; ++c) {
        const int cur = c & 1;
        const int k0  = c * KC;

        if (c + 1 < nch) {
            const int nk0 = k0 + KC;
#pragma unroll
            for (int r = 0; r < 2; ++r) {
                const int idx  = (r == 0) ? idx0 : idx1;
                const int fl   = idx >> 4;
                const int kloc = (idx & 15) << 2;
                async_copy_b128(lds_off(&rawQ[cur ^ 1][idx * 4]),
                                w2q + (size_t)(h0 + fl) * F_DIM + nk0 + kloc);
            }
            wait_async_le2();
        } else {
            wait_async_le0();
        }
        __syncthreads();

#pragma unroll
        for (int r = 0; r < 2; ++r) {
            const int idx  = (r == 0) ? idx0 : idx1;
            const int fl   = idx >> 4;
            const int kloc = (idx & 15) << 2;
            const size_t soff = (size_t)g * F_DIM + k0 + kloc;
            dequant4(&ldsB[fl * LBST + kloc], &rawQ[cur][idx * 4],
                     w2s + soff, w2z + soff);
        }
        __syncthreads();

#pragma unroll
        for (int kk = 0; kk < KC; kk += 32) {
            v16h b = load_b_lds(ldsB, wc * 16, kk, lane);
#pragma unroll
            for (int t = 0; t < NTT; ++t) {
                v16h a = load_a_global(inter, F_DIM, tw + t * 16, k0 + kk, lane);
                acc[t] = wmma_f16(a, b, acc[t]);
            }
        }
    }

    const int n     = lane & 15;
    const int mbase = (lane < 16) ? 0 : 8;
#pragma unroll
    for (int t = 0; t < NTT; ++t) {
#pragma unroll
        for (int i = 0; i < 8; ++i) {
            int trow = tw + t * 16 + mbase + i;
            out[(size_t)trow * H_DIM + h0 + wc * 16 + n] = acc[t][i];
        }
    }
}

// ===========================================================================
extern "C" void kernel_launch(void* const* d_in, const int* in_sizes, int n_in,
                              void* d_out, int out_size, void* d_ws, size_t ws_size,
                              hipStream_t stream) {
    const float* x   = (const float*)d_in[0];
    const int*   w1q = (const int*)d_in[1];
    const float* w1s = (const float*)d_in[2];
    const float* w1z = (const float*)d_in[3];
    const int*   w2q = (const int*)d_in[4];
    const float* w2s = (const float*)d_in[5];
    const float* w2z = (const float*)d_in[6];
    const int*   w3q = (const int*)d_in[7];
    const float* w3s = (const float*)d_in[8];
    const float* w3z = (const float*)d_in[9];

    // workspace: inter [T*F] f16, then xh [T*H] f16 (~18.9 MB)
    _Float16* inter = (_Float16*)d_ws;
    _Float16* xh    = inter + (size_t)T_DIM * F_DIM;
    float*    out   = (float*)d_out;

    dim3 blk(256, 1, 1);

    cvt_x_kernel<<<dim3((T_DIM * H_DIM) / (256 * 4), 1, 1), blk, 0, stream>>>(x, xh);

    hqq_gateup_kernel<<<dim3(F_DIM / FT, 1, 1), blk, 0, stream>>>(
        xh, w1q, w1s, w1z, w3q, w3s, w3z, inter);

    hqq_down_kernel<<<dim3(H_DIM / FT, 1, 1), blk, 0, stream>>>(
        inter, w2q, w2s, w2z, out);
}
